// VQ_14396730376270
// MI455X (gfx1250) — matrix operand order repaired and checked by reading, not compiled, not run on hardware
//
#include <hip/hip_runtime.h>
#include <hip/hip_bf16.h>

// ---------------------------------------------------------------------------
// VQ-VAE vector quantization for MI455X (gfx1250).
//   x:      [16, 512, 64, 64] f32      weight: [2048, 512] f32
//   out:    z_q [16,512,64,64] f32  ++ commitment_loss ++ codebook_loss
//
// Distance GEMM: v_wmma_f32_16x16x32_bf16. Argmin depends only on
// -2 x.w + |w|^2 (|x|^2 drops out), which bf16 inputs + f32 accumulate
// resolve comfortably. Codebook tiles are staged into LDS by the Tensor
// Data Mover (tensor_load_to_lds, double-buffered, TDM row padding for
// bank-conflict-free ds_read_b128) and shared by 16 waves per block.
// B fragments are loaded two chunks at a time (both live across a WMMA so
// the RA cannot coalesce them), relaxing per-chunk waits to dscnt<=2.
// ---------------------------------------------------------------------------

typedef __attribute__((ext_vector_type(16))) __bf16 v16bf;
typedef __attribute__((ext_vector_type(8)))  __bf16 bf16x8;
typedef __attribute__((ext_vector_type(8)))  float  v8f;
typedef __attribute__((ext_vector_type(4)))  unsigned int v4u;
typedef __attribute__((ext_vector_type(8)))  int    v8i;
typedef __attribute__((ext_vector_type(4)))  int    v4i;

static constexpr int NCODES = 2048;
static constexpr int CDIM   = 512;
static constexpr int BATCH  = 16;
static constexpr int HW     = 64 * 64;            // 4096 spatial positions
static constexpr int NROWS  = BATCH * HW;         // 65536 vectors
static constexpr int WAVES  = 16;                 // waves per block
static constexpr int MB     = WAVES * 16;         // 256 rows per block
static constexpr int BSTR   = 520;                // B tile row stride in bf16 (1040 B)
static constexpr int NT     = NCODES / 16;        // 128 code tiles

// --- 1. codebook prep: f32 -> bf16 copy + |w_k|^2 ---------------------------
__global__ void vq_prep(const float* __restrict__ w,
                        __bf16* __restrict__ wb,
                        float* __restrict__ wsq) {
    const int k = blockIdx.x;
    const float* row = w + (size_t)k * CDIM;
    __bf16* out = wb + (size_t)k * CDIM;
    float acc = 0.f;
    for (int c = threadIdx.x; c < CDIM; c += blockDim.x) {
        float v = row[c];
        out[c] = (__bf16)v;
        acc += v * v;
    }
    for (int off = 16; off; off >>= 1) acc += __shfl_xor(acc, off, 32);
    __shared__ float part[8];
    if ((threadIdx.x & 31) == 0) part[threadIdx.x >> 5] = acc;
    __syncthreads();
    if (threadIdx.x == 0) {
        float s = 0.f;
        for (int i = 0; i < 8; ++i) s += part[i];
        wsq[k] = s;
    }
}

// --- TDM: async-load one 16x512 bf16 codebook tile into LDS -----------------
// D# per ISA ch.8: 2-D tile, data_size=2B, row padding 16B after each 1024B
// so LDS rows land on a 1040B stride (bank-conflict-free b128 reads).
// This toolchain's builtin takes 6 args; the 5th (int32x8) corresponds to the
// unused VADDR4 slot in the VIMAGE encoding -> zero.
__device__ __forceinline__ void tdm_load_tile(const __bf16* gsrc, unsigned int lds_off) {
    const unsigned long long ga = (unsigned long long)(uintptr_t)gsrc;
    v4u g0;
    g0[0] = 1u;                                                 // count=1 (valid user D#)
    g0[1] = lds_off;                                            // lds_addr (bytes)
    g0[2] = (unsigned int)ga;                                   // global_addr[31:0]
    g0[3] = (unsigned int)((ga >> 32) & 0x01FFFFFFu)            // global_addr[56:32]
          | (2u << 30);                                         // type=2 ("image")
    v8i g1;
    g1[0] = (int)((1u << 16)        // data_size = 1 -> 2 bytes
                | (1u << 20)        // pad_enable
                | (7u << 22)        // pad_interval: 256 dwords = 1024 B
                | (3u << 25));      // pad_amount:   4 dwords = 16 B
    g1[1] = (int)(512u  << 16);     // tensor_dim0 = 512 elements
    g1[2] = (int)(2048u << 16);     // tensor_dim1 = 2048 rows
    g1[3] = (int)(512u  << 16);     // tile_dim0 = 512
    g1[4] = 16;                     // tile_dim1 = 16 rows
    g1[5] = 512;                    // tensor_dim0_stride = 512 elements
    g1[6] = 0;
    g1[7] = 0;
    const v4i gz  = {0, 0, 0, 0};                   // groups 2/3 unused (2-D tile)
    const v8i gz8 = {0, 0, 0, 0, 0, 0, 0, 0};       // unused VADDR4 slot
    __builtin_amdgcn_tensor_load_to_lds(g0, g1, gz, gz, gz8, 0);
}

// --- 2. WMMA distance GEMM + running argmin ---------------------------------
__global__ __launch_bounds__(WAVES * 32)
void vq_argmin(const float* __restrict__ x,
               const __bf16* __restrict__ wb,
               const float* __restrict__ wsq,
               int* __restrict__ amin) {
    __shared__ __align__(16) __bf16 sB[2][16 * BSTR];   // double-buffered B tile (~33 KB)

    const int tid   = threadIdx.x;
    const int wave  = tid >> 5;
    const int lane  = tid & 31;
    const int lrow  = lane & 15;               // A: M row / B,C: N column
    const int lhalf = lane >> 4;

    const int row0  = blockIdx.x * MB;         // MB divides HW: one batch image
    const int batch = row0 >> 12;              // /4096
    const int hw0   = row0 & (HW - 1);

    // ---- A: load this lane's x row straight from global (NCHW strided),
    // converting to bf16 into WMMA A fragments. Done once per block.
    // A element e -> K = 32*kc + 16*lhalf + e.
    union AB { v16bf v; bf16x8 h[2]; };
    const int hw = hw0 + wave * 16 + lrow;
    const float* xcol = x + (size_t)batch * CDIM * HW + hw;
    AB a[16];
    #pragma unroll
    for (int kc = 0; kc < 16; ++kc) {
        #pragma unroll
        for (int e = 0; e < 16; ++e) {
            const int c = kc * 32 + 16 * lhalf + e;
            a[kc].v[e] = (__bf16)xcol[(size_t)c * HW];
        }
    }

    const unsigned int ldsB[2] = {
        (unsigned int)(uintptr_t)&sB[0][0],    // generic->LDS: low 32 bits = offset
        (unsigned int)(uintptr_t)&sB[1][0]
    };

    // Prologue: TDM fetch of tile 0.
    if (wave == 0) {
        tdm_load_tile(wb, ldsB[0]);
        __builtin_amdgcn_s_wait_tensorcnt(0);
    }
    __syncthreads();

    float best[8];
    int   bidx[8];
    #pragma unroll
    for (int j = 0; j < 8; ++j) { best[j] = -3.4e38f; bidx[j] = 0; }

    for (int nt = 0; nt < NT; ++nt) {
        // Kick off next tile while computing this one.
        if (wave == 0 && nt + 1 < NT)
            tdm_load_tile(wb + (size_t)(nt + 1) * 16 * CDIM, ldsB[(nt + 1) & 1]);

        const int code = nt * 16 + lrow;
        const float t0 = -0.5f * wsq[code];    // issue early; used after the GEMM

        // B element e -> K = 32*kc + 16*lhalf + e: contiguous 16 bf16 per lane.
        // Row stride 1040 B => lane lrow starts at bank 4*lrow: conflict-free.
        const __bf16* brow = &sB[nt & 1][lrow * BSTR + 16 * lhalf];

        // Two chunks in flight: b1 is live across the first WMMA, so the RA
        // must keep two distinct fragment register pairs; the wait before the
        // first WMMA relaxes to dscnt<=2 (in-order LDS returns).
        v8f acc = {};
        #pragma unroll
        for (int kc = 0; kc < 16; kc += 2) {
            AB b0, b1;
            b0.h[0] = *reinterpret_cast<const bf16x8*>(brow + kc * 32);
            b0.h[1] = *reinterpret_cast<const bf16x8*>(brow + kc * 32 + 8);
            b1.h[0] = *reinterpret_cast<const bf16x8*>(brow + kc * 32 + 32);
            b1.h[1] = *reinterpret_cast<const bf16x8*>(brow + kc * 32 + 40);
            acc = __builtin_amdgcn_wmma_f32_16x16x32_bf16(
                      false, a[kc].v, false, b0.v, (short)0, acc, false, false);
            acc = __builtin_amdgcn_wmma_f32_16x16x32_bf16(
                      false, a[kc + 1].v, false, b1.v, (short)0, acc, false, false);
        }

        // maximize  x.w - 0.5*|w|^2  ==  minimize squared distance
        #pragma unroll
        for (int j = 0; j < 8; ++j) {
            const float t = acc[j] + t0;       // C layout: VGPR j -> M = j + 8*lhalf
            if (t > best[j]) { best[j] = t; bidx[j] = code; }   // earliest index on ties
        }

        if (wave == 0) __builtin_amdgcn_s_wait_tensorcnt(0);
        __syncthreads();   // next tile ready; everyone done with current buffer
    }

    // Arg-reduce across the 16 lanes of each half (lane saw codes == lrow mod 16).
    #pragma unroll
    for (int j = 0; j < 8; ++j) {
        float v = best[j];
        int   i = bidx[j];
        #pragma unroll
        for (int off = 1; off < 16; off <<= 1) {
            const float ov = __shfl_xor(v, off, 32);
            const int   oi = __shfl_xor(i, off, 32);
            if (ov > v || (ov == v && oi < i)) { v = ov; i = oi; }
        }
        if (lrow == 0)
            amin[row0 + wave * 16 + (j + 8 * lhalf)] = i;
    }
}

// --- 3. gather z_q (NCHW) + sum (x - z_q)^2 ---------------------------------
__global__ void vq_gather_loss(const float* __restrict__ x,
                               const float* __restrict__ w,
                               const int* __restrict__ amin,
                               float* __restrict__ zq,
                               double* __restrict__ loss) {
    const size_t base = (size_t)blockIdx.x * blockDim.x * 8 + threadIdx.x;
    float acc = 0.f;
    #pragma unroll
    for (int e = 0; e < 8; ++e) {
        const size_t idx = base + (size_t)e * blockDim.x;     // grid sized exactly
        const int hw = (int)(idx & (HW - 1));                 // bits [11:0]
        const int c  = (int)((idx >> 12) & (CDIM - 1));       // bits [20:12]
        const int b  = (int)(idx >> 21);                      // bits [24:21]
        const int k  = amin[(b << 12) + hw];
        const float wv = w[(size_t)k * CDIM + c];             // 4MB codebook lives in L2
        zq[idx] = wv;
        const float d = x[idx] - wv;
        acc += d * d;
    }
    for (int off = 16; off; off >>= 1) acc += __shfl_xor(acc, off, 32);
    __shared__ float part[8];
    if ((threadIdx.x & 31) == 0) part[threadIdx.x >> 5] = acc;
    __syncthreads();
    if (threadIdx.x == 0) {
        float s = 0.f;
        for (int i = 0; i < 8; ++i) s += part[i];
        atomicAdd(loss, (double)s);                            // global_atomic_add_f64
    }
}

// --- 4. finalize scalars -----------------------------------------------------
__global__ void vq_final(const double* __restrict__ loss, float* __restrict__ out2) {
    const double mean = *loss / (double)((size_t)BATCH * CDIM * HW);
    out2[0] = (float)(0.25 * mean);   // commitment_loss
    out2[1] = (float)mean;            // codebook_loss
}

// ---------------------------------------------------------------------------
extern "C" void kernel_launch(void* const* d_in, const int* in_sizes, int n_in,
                              void* d_out, int out_size, void* d_ws, size_t ws_size,
                              hipStream_t stream) {
    (void)in_sizes; (void)n_in; (void)out_size; (void)ws_size;
    const float* x = (const float*)d_in[0];
    const float* w = (const float*)d_in[1];
    float* out = (float*)d_out;

    char* ws = (char*)d_ws;
    __bf16* wb   = (__bf16*)ws;                                   // 2 MB bf16 codebook
    float*  wsq  = (float*)(ws + (size_t)2 * 1024 * 1024);        // 8 KB |w|^2
    int*    amin = (int*)(ws + (size_t)2 * 1024 * 1024 + 8192);   // 256 KB argmin
    double* loss = (double*)(ws + (size_t)2 * 1024 * 1024 + 8192 + 262144);

    (void)hipMemsetAsync(loss, 0, sizeof(double), stream);
    vq_prep<<<NCODES, 256, 0, stream>>>(w, wb, wsq);
    vq_argmin<<<NROWS / MB, WAVES * 32, 0, stream>>>(x, wb, wsq, amin);

    const size_t total = (size_t)BATCH * CDIM * HW;               // 33,554,432
    vq_gather_loss<<<(int)(total / (256 * 8)), 256, 0, stream>>>(x, w, amin, out, loss);
    vq_final<<<1, 1, 0, stream>>>(loss, out + total);
}
